// BinaryConv2dLayer_68590627717440
// MI455X (gfx1250) — matrix operand order repaired and checked by reading, not compiled, not run on hardware
//
#include <hip/hip_runtime.h>
#include <stdint.h>

typedef __attribute__((ext_vector_type(16))) __bf16       v16bf;
typedef __attribute__((ext_vector_type(8)))  float        v8f;
typedef __attribute__((ext_vector_type(8)))  unsigned int v8u;
typedef __attribute__((ext_vector_type(4)))  float        v4f;

#define HH     256
#define WW     256
#define CI     64
#define CO     64
#define NIMG   16
#define KSTEPS 18            // 576 / 32
#define NFRAG  (KSTEPS * 4)  // 72 B-fragments (4 co-tiles per k-step)
#define WPACK_V8U   (NFRAG * 32)       // 2304 x 32B fragment-lanes
#define WPACK_BYTES (WPACK_V8U * 32)   // 73728 B
#define WPACK_B128  (WPACK_BYTES / 16) // 4608 x 16B async-copy units

// ---------------------------------------------------------------------------
// Kernel 1: scale = sum(W) / sum(sign(W))   (single workgroup reduction)
// ---------------------------------------------------------------------------
__global__ void bconv_scale_kernel(const float* __restrict__ Wt,
                                   float* __restrict__ scale_out) {
    __shared__ float ssum[256];
    __shared__ float ssgn[256];
    const int tid = threadIdx.x;
    float s = 0.f, g = 0.f;
    for (int i = tid; i < 3 * 3 * 64 * 64; i += 256) {
        const float w = Wt[i];
        s += w;
        g += (w > 0.f ? 1.f : (w < 0.f ? -1.f : 0.f));
    }
    ssum[tid] = s;
    ssgn[tid] = g;
    __syncthreads();
    for (int off = 128; off > 0; off >>= 1) {
        if (tid < off) {
            ssum[tid] += ssum[tid + off];
            ssgn[tid] += ssgn[tid + off];
        }
        __syncthreads();
    }
    if (tid == 0) scale_out[0] = ssum[0] / ssgn[0];
}

// ---------------------------------------------------------------------------
// Kernel 2: binarize + pack W into WMMA B-fragment order (bf16).
// Layout: wpack[((kstep*4 + co_tile)*32 + lane)*16 + i]
// B (32x16, 16-bit, wave32): lane holds col n = lane%16,
//   K = kstep*32 + 16*(lane>>4) + i, i = 0..15 contiguous.
// W flat HWIO index for GEMM-k and col n:  W[k*64 + n]
// ---------------------------------------------------------------------------
__global__ void bconv_pack_kernel(const float* __restrict__ Wt,
                                  const float* __restrict__ scale_ptr,
                                  __bf16* __restrict__ wpack) {
    const int tid = blockIdx.x * blockDim.x + threadIdx.x;
    if (tid >= NFRAG * 32) return;
    const float scale = scale_ptr[0];
    const int lane  = tid & 31;
    const int frag  = tid >> 5;   // kstep*4 + co_tile
    const int kstep = frag >> 2;
    const int t     = frag & 3;
    const int n  = t * 16 + (lane & 15);
    const int kb = kstep * 32 + ((lane >> 4) * 16);
#pragma unroll
    for (int i = 0; i < 16; ++i) {
        const float w  = Wt[(kb + i) * 64 + n];
        const float sg = (w > 0.f ? 1.f : (w < 0.f ? -1.f : 0.f));
        wpack[tid * 16 + i] = (__bf16)(sg * scale);
    }
}

// ---------------------------------------------------------------------------
// A-fragment gather: 16 bf16 per lane from one 16-float contiguous run.
// Lanes 0-15 hold K {c0..c0+15}; lanes 16-31 are offset +16 (koff).
// ---------------------------------------------------------------------------
__device__ __forceinline__ v16bf bconv_load_a(const float* __restrict__ xp,
                                              int c0, bool valid) {
    v16bf a = {};
    if (valid) {
        const v4f f0 = *(const v4f*)(xp + c0);
        const v4f f1 = *(const v4f*)(xp + c0 + 4);
        const v4f f2 = *(const v4f*)(xp + c0 + 8);
        const v4f f3 = *(const v4f*)(xp + c0 + 12);
#pragma unroll
        for (int i = 0; i < 4; ++i) {
            a[i]      = (__bf16)f0[i];
            a[i + 4]  = (__bf16)f1[i];
            a[i + 8]  = (__bf16)f2[i];
            a[i + 12] = (__bf16)f3[i];
        }
    }
    return a;
}

// ---------------------------------------------------------------------------
// Kernel 3: implicit-GEMM conv. 8 waves / block, wave = 32 rows x 64 co.
// Block = 256 rows = one full image row (shared n,h). B staged in LDS via
// gfx1250 async global->LDS DMA (ASYNCcnt domain), then read via ds_load_b128.
// ---------------------------------------------------------------------------
__global__ __launch_bounds__(256) void bconv_wmma_kernel(
    const float* __restrict__ x,
    const __bf16* __restrict__ wpack,
    float* __restrict__ out) {

    __shared__ v8u lds_b[WPACK_V8U];   // 73728 B packed weights

    const int lane = threadIdx.x & 31;
    const int wave = threadIdx.x >> 5;

    // Async stage of packed weights into LDS (once per block): 4608 x 16B,
    // striped across 256 threads. Data goes straight to LDS (no VGPRs),
    // tracked with ASYNCcnt.
    {
        const uint32_t lds_base = (uint32_t)(uintptr_t)(&lds_b[0]);  // low 32b = LDS offset
        uint64_t       gaddr    = (uint64_t)(uintptr_t)wpack + (uint64_t)threadIdx.x * 16u;
        uint32_t       laddr    = lds_base + (uint32_t)threadIdx.x * 16u;
#pragma unroll
        for (int it = 0; it < WPACK_B128 / 256; ++it) {              // 18 iterations
            asm volatile("global_load_async_to_lds_b128 %0, %1, off"
                         :: "v"(laddr), "v"(gaddr)
                         : "memory");
            laddr += 256u * 16u;
            gaddr += 256u * 16u;
        }
        asm volatile("s_wait_asynccnt 0x0" ::: "memory");
    }
    __syncthreads();

    const int mbase = blockIdx.x * 256 + wave * 32;   // 32 rows per wave
    const int nimg  = mbase >> 16;                    // 256*256 rows per image
    const int hrow  = (mbase >> 8) & 255;
    const int w0    = mbase & 255;                    // wave*32
    const int wcol0 = w0 + (lane & 15);
    const int wcol1 = wcol0 + 16;
    const int koff  = (lane >> 4) * 16;               // A-layout K offset

    v8f acc0 = {}, acc1 = {}, acc2 = {}, acc3 = {};   // rows mbase..+15
    v8f acc4 = {}, acc5 = {}, acc6 = {}, acc7 = {};   // rows mbase+16..+31

    for (int pos = 0; pos < 9; ++pos) {               // 3x3 taps
        const int kh   = pos / 3;
        const int kw   = pos - kh * 3;
        const int h_in = hrow + kh - 1;
        if ((unsigned)h_in >= (unsigned)HH) continue; // uniform: skip tap
        const int  w_in0   = wcol0 + kw - 1;
        const int  w_in1   = wcol1 + kw - 1;
        const bool wvalid0 = ((unsigned)w_in0 < (unsigned)WW);
        const bool wvalid1 = ((unsigned)w_in1 < (unsigned)WW);
        const float* __restrict__ xrow =
            x + (((size_t)nimg * HH + (size_t)h_in) * WW) * CI;
        const float* __restrict__ xp0 = xrow + (size_t)(wvalid0 ? w_in0 : 0) * CI;
        const float* __restrict__ xp1 = xrow + (size_t)(wvalid1 ? w_in1 : 0) * CI;

#pragma unroll
        for (int half = 0; half < 2; ++half) {        // 2 x 32 input channels
            const int c0 = half * 32 + koff;
            const v16bf a0 = bconv_load_a(xp0, c0, wvalid0);
            const v16bf a1 = bconv_load_a(xp1, c0, wvalid1);

            const int kstep = pos * 2 + half;
            const v16bf b0 = __builtin_bit_cast(v16bf, lds_b[(kstep * 4 + 0) * 32 + lane]);
            const v16bf b1 = __builtin_bit_cast(v16bf, lds_b[(kstep * 4 + 1) * 32 + lane]);
            const v16bf b2 = __builtin_bit_cast(v16bf, lds_b[(kstep * 4 + 2) * 32 + lane]);
            const v16bf b3 = __builtin_bit_cast(v16bf, lds_b[(kstep * 4 + 3) * 32 + lane]);

            acc0 = __builtin_amdgcn_wmma_f32_16x16x32_bf16(false, a0, false, b0,
                                                           (short)0, acc0, false, false);
            acc1 = __builtin_amdgcn_wmma_f32_16x16x32_bf16(false, a0, false, b1,
                                                           (short)0, acc1, false, false);
            acc2 = __builtin_amdgcn_wmma_f32_16x16x32_bf16(false, a0, false, b2,
                                                           (short)0, acc2, false, false);
            acc3 = __builtin_amdgcn_wmma_f32_16x16x32_bf16(false, a0, false, b3,
                                                           (short)0, acc3, false, false);
            acc4 = __builtin_amdgcn_wmma_f32_16x16x32_bf16(false, a1, false, b0,
                                                           (short)0, acc4, false, false);
            acc5 = __builtin_amdgcn_wmma_f32_16x16x32_bf16(false, a1, false, b1,
                                                           (short)0, acc5, false, false);
            acc6 = __builtin_amdgcn_wmma_f32_16x16x32_bf16(false, a1, false, b2,
                                                           (short)0, acc6, false, false);
            acc7 = __builtin_amdgcn_wmma_f32_16x16x32_bf16(false, a1, false, b3,
                                                           (short)0, acc7, false, false);
        }
    }

    // C/D layout: lane holds col n = lane%16; VGPR r holds row r + 8*(lane>>4).
    const int mloc_base = (lane >> 4) * 8;
    const int co        = lane & 15;
#pragma unroll
    for (int r = 0; r < 8; ++r) {
        const size_t o0 = ((size_t)(mbase + r + mloc_base)) * CO + (size_t)co;
        const size_t o1 = o0 + (size_t)16 * CO;
        float v;
        v = acc0[r]; out[o0 +  0] = v > 0.f ? v : 0.f;
        v = acc1[r]; out[o0 + 16] = v > 0.f ? v : 0.f;
        v = acc2[r]; out[o0 + 32] = v > 0.f ? v : 0.f;
        v = acc3[r]; out[o0 + 48] = v > 0.f ? v : 0.f;
        v = acc4[r]; out[o1 +  0] = v > 0.f ? v : 0.f;
        v = acc5[r]; out[o1 + 16] = v > 0.f ? v : 0.f;
        v = acc6[r]; out[o1 + 32] = v > 0.f ? v : 0.f;
        v = acc7[r]; out[o1 + 48] = v > 0.f ? v : 0.f;
    }
}

// ---------------------------------------------------------------------------
extern "C" void kernel_launch(void* const* d_in, const int* in_sizes, int n_in,
                              void* d_out, int out_size, void* d_ws, size_t ws_size,
                              hipStream_t stream) {
    (void)in_sizes; (void)n_in; (void)out_size; (void)ws_size;
    const float* x  = (const float*)d_in[0];
    const float* Wt = (const float*)d_in[1];
    float* out      = (float*)d_out;

    __bf16* wpack = (__bf16*)d_ws;                         // 73728 B
    float*  scale = (float*)((char*)d_ws + WPACK_BYTES);   // 4 B

    bconv_scale_kernel<<<1, 256, 0, stream>>>(Wt, scale);
    bconv_pack_kernel<<<(NFRAG * 32 + 255) / 256, 256, 0, stream>>>(Wt, scale, wpack);

    const int nblocks = NIMG * HH;                         // 4096, one image row each
    bconv_wmma_kernel<<<nblocks, 256, 0, stream>>>(x, wpack, out);
}
